// SpectralEquivariantGNN_74071005987606
// MI455X (gfx1250) — matrix operand order repaired and checked by reading, not compiled, not run on hardware
//
#include <hip/hip_runtime.h>
#include <math.h>

// ---------------------------------------------------------------------------
// SpectralEquivariantGNN forward for MI455X (gfx1250, wave32, WMMA)
// GEMMs: v_wmma_f32_16x16x32_f16, f32 accumulate.
//  - Weights pre-swizzled to f16 in the exact WMMA B-fragment layout
//    [k-step][16col-tile][lane][16 halves]  -> B fragment = 2x b128 loads,
//    no cvts, immediate-offset addressing.
//  - A read f32 (4x b128) and packed to f16 with v_cvt_pk_f16_f32.
//  - Ragged-K embedding inputs zero-padded to K=32 once.
// ---------------------------------------------------------------------------

typedef __attribute__((ext_vector_type(16))) _Float16 v16h;
typedef __attribute__((ext_vector_type(8)))  _Float16 v8h;
typedef __attribute__((ext_vector_type(8)))  float    v8f;
typedef __attribute__((ext_vector_type(4)))  float    v4f;

#define NNODE 16384
#define BG    32
#define NPGc  512
#define EEDGE 262144
#define DDIM  128
#define HHEAD 4
#define KQ    64
#define ETOT  (EEDGE + NNODE)

// ---------------- activation helpers ----------------
template<int ACT>
__device__ __forceinline__ float actf(float v) {
  if (ACT == 1) return v / (1.f + expf(-v));                                   // SiLU
  if (ACT == 2) return 0.5f * v * (1.f + erff(v * 0.70710678118654752f));      // exact GELU
  return v;
}
__device__ __forceinline__ float geluf(float v) {
  return 0.5f * v * (1.f + erff(v * 0.70710678118654752f));
}
__device__ __forceinline__ float softplusf(float x) {
  return x > 0.f ? x + log1pf(expf(-x)) : log1pf(expf(x));
}

// monotone-int encoding for float atomic max
__device__ __forceinline__ void atomicMaxF(unsigned* a, float f) {
  unsigned v = __float_as_uint(f);
  unsigned e = (v & 0x80000000u) ? ~v : (v | 0x80000000u);
  atomicMax(a, e);
}
__device__ __forceinline__ float decMaxF(unsigned e) {
  unsigned v = (e & 0x80000000u) ? (e & 0x7FFFFFFFu) : ~e;
  return __uint_as_float(v);
}

// ---------------------------------------------------------------------------
// Weight swizzle: W[K,N] f32 (leading dim ldw) -> f16 fragment-ordered buffer
//   out[(ks*(N/16)+tt)*512 + lane*16 + e] = W[ks*32 + ((lane&16)?8:0) + (e<8 ? e : e+8),
//                                             tt*16 + (lane&15)]
// so a wave's B fragment for (k-step ks, 16-col tile tt) is the contiguous
// 32B chunk at lane*16.
// ---------------------------------------------------------------------------
__global__ void swizzle_w_kernel(const float* __restrict__ W, int ldw, int N, int K,
                                 _Float16* __restrict__ out) {
  int i = blockIdx.x * blockDim.x + threadIdx.x;
  if (i >= K * N) return;
  const int e    = i & 15;
  const int lane = (i >> 4) & 31;
  const int nTt  = N >> 4;
  const int tt   = (i >> 9) % nTt;
  const int ks   = (i >> 9) / nTt;
  const int krow = ks * 32 + ((lane & 16) ? 8 : 0) + (e < 8 ? e : e + 8);
  const int col  = tt * 16 + (lane & 15);
  out[i] = (_Float16)W[(size_t)krow * ldw + col];
}

// ---------------------------------------------------------------------------
// Guard-free WMMA GEMM:  C[M,N] = act(A[M,K] @ W[K,N] + bias[N])
// wave32; each wave computes a 16x64 strip: 4 x v_wmma_f32_16x16x32_f16 per
// K-step, A fragment reused 4x.  Requires M%16==0, N%64==0, K%32==0,
// lda%4==0.  Wsw is the pre-swizzled f16 weight buffer (see above).
// ---------------------------------------------------------------------------
__device__ __forceinline__ v16h ldfrag(const _Float16* p) {
  v8h lo = *(const v8h*)p;
  v8h hi = *(const v8h*)(p + 8);
  return __builtin_shufflevector(lo, hi, 0,1,2,3,4,5,6,7,8,9,10,11,12,13,14,15);
}

template<int ACT>
__global__ __launch_bounds__(256) void wmma_gemm_fast(
    const float* __restrict__ A, int lda,
    const _Float16* __restrict__ Wsw,
    const float* __restrict__ bias,
    float* __restrict__ C, int ldc,
    int M, int N, int K)
{
  const int warp = (int)((blockIdx.x * blockDim.x + threadIdx.x) >> 5);
  const int lane = threadIdx.x & 31;
  const int nTn  = N >> 6;                 // 64-wide column groups
  const int tm   = warp / nTn;
  const int tn   = warp % nTn;
  if (tm * 16 >= M) return;

  const int row = tm * 16 + (lane & 15);
  const int ka0 = (lane & 16) ? 8 : 0;     // A: elems 0..7 -> K=ka0..ka0+7, 8..15 -> +16
  const int col = tn * 64 + (lane & 15);
  const float* __restrict__ Arow = A + (size_t)row * lda;
  const _Float16* __restrict__ wp = Wsw + ((size_t)tn * 4) * 512 + lane * 16;
  const size_t wstep = (size_t)(N >> 4) * 512;   // halves per k-step

  v8f acc0 = {}, acc1 = {}, acc2 = {}, acc3 = {};

  for (int k0 = 0; k0 < K; k0 += 32) {
    __builtin_prefetch(Arow + k0 + 128, 0, 1);      // speculative, no guard
    // A fragment: 4x 16B vector loads + packed f16 conversion
    const v4f A0 = *(const v4f*)(Arow + k0 + ka0);
    const v4f A1 = *(const v4f*)(Arow + k0 + ka0 + 4);
    const v4f A2 = *(const v4f*)(Arow + k0 + ka0 + 16);
    const v4f A3 = *(const v4f*)(Arow + k0 + ka0 + 20);
    v16h a;
#pragma unroll
    for (int e = 0; e < 4; ++e) {
      a[e]      = (_Float16)A0[e];
      a[e + 4]  = (_Float16)A1[e];
      a[e + 8]  = (_Float16)A2[e];
      a[e + 12] = (_Float16)A3[e];
    }
    // B fragments: 8x b128, immediate offsets off one base pointer
    const v16h b0 = ldfrag(wp);
    const v16h b1 = ldfrag(wp + 512);
    const v16h b2 = ldfrag(wp + 1024);
    const v16h b3 = ldfrag(wp + 1536);
    wp += wstep;

    acc0 = __builtin_amdgcn_wmma_f32_16x16x32_f16(false, a, false, b0, (short)0, acc0, false, false);
    acc1 = __builtin_amdgcn_wmma_f32_16x16x32_f16(false, a, false, b1, (short)0, acc1, false, false);
    acc2 = __builtin_amdgcn_wmma_f32_16x16x32_f16(false, a, false, b2, (short)0, acc2, false, false);
    acc3 = __builtin_amdgcn_wmma_f32_16x16x32_f16(false, a, false, b3, (short)0, acc3, false, false);
  }

  const int rowAdd = (lane & 16) ? 8 : 0;
  const float bv0 = bias ? bias[col +  0] : 0.f;
  const float bv1 = bias ? bias[col + 16] : 0.f;
  const float bv2 = bias ? bias[col + 32] : 0.f;
  const float bv3 = bias ? bias[col + 48] : 0.f;
#pragma unroll
  for (int r = 0; r < 8; ++r) {
    const int rr = tm * 16 + rowAdd + r;
    float* cr = C + (size_t)rr * ldc + col;
    cr[0]  = actf<ACT>(acc0[r] + bv0);
    cr[16] = actf<ACT>(acc1[r] + bv1);
    cr[32] = actf<ACT>(acc2[r] + bv2);
    cr[48] = actf<ACT>(acc3[r] + bv3);
  }
}

// ---------------- K-padding for the two ragged-K embedding GEMMs ----------------
__global__ void pad_a_kernel(const float* __restrict__ src, int Kin,
                             float* __restrict__ dst, long rows) {
  long i = blockIdx.x * (long)blockDim.x + threadIdx.x;
  if (i >= rows * 32) return;
  const int c = (int)(i & 31); const long r = i >> 5;
  dst[i] = (c < Kin) ? src[r * Kin + c] : 0.f;
}
__global__ void pad_w_kernel(const float* __restrict__ w, int Kin, int N,
                             float* __restrict__ dst) {
  int i = blockIdx.x * blockDim.x + threadIdx.x;
  if (i >= 32 * N) return;
  const int r = i / N, c = i - r * N;
  dst[i] = (r < Kin) ? w[r * N + c] : 0.f;
}

// ---------------------------------------------------------------------------
// LayerNorm over D=128 (one block per row).  y = LN(x + res? + bias?)*g + beta
// ---------------------------------------------------------------------------
__global__ void ln_kernel(const float* __restrict__ x, const float* __restrict__ res,
                          const float* __restrict__ bias,
                          const float* __restrict__ g, const float* __restrict__ beta,
                          float* __restrict__ y)
{
  __shared__ float red[128];
  const int row = blockIdx.x, d = threadIdx.x;
  float v = x[(size_t)row * 128 + d];
  if (res)  v += res[(size_t)row * 128 + d];
  if (bias) v += bias[d];
  red[d] = v; __syncthreads();
  for (int o = 64; o > 0; o >>= 1) { if (d < o) red[d] += red[d + o]; __syncthreads(); }
  const float mean = red[0] * (1.f / 128.f); __syncthreads();
  const float c = v - mean;
  red[d] = c * c; __syncthreads();
  for (int o = 64; o > 0; o >>= 1) { if (d < o) red[d] += red[d + o]; __syncthreads(); }
  const float var = red[0] * (1.f / 128.f);
  y[(size_t)row * 128 + d] = c * rsqrtf(var + 1e-5f) * g[d] + beta[d];
}

__global__ void fill_u32_kernel(unsigned* p, unsigned v, long n) {
  long i = blockIdx.x * (long)blockDim.x + threadIdx.x;
  if (i < n) p[i] = v;
}
__global__ void gelu_add_kernel(float* __restrict__ h, const float* __restrict__ x, long n) {
  long i = blockIdx.x * (long)blockDim.x + threadIdx.x;
  if (i < n) h[i] += geluf(x[i]);
}
__global__ void bcast_query_kernel(const float* __restrict__ q, float* __restrict__ t, long n) {
  long i = blockIdx.x * (long)blockDim.x + threadIdx.x;
  if (i < n) t[i] = q[i % (KQ * 128)];
}

// ---------------- GAT self-loop edge features ----------------
__global__ void eloop_acc_kernel(const float* __restrict__ e, const int* __restrict__ dst,
                                 float* __restrict__ acc, float* __restrict__ cnt) {
  const int ed = blockIdx.x, d = threadIdx.x;
  const int dd = dst[ed];
  atomicAdd(&acc[(size_t)dd * 128 + d], e[(size_t)ed * 128 + d]);
  if (d == 0) atomicAdd(&cnt[dd], 1.f);
}
__global__ void eloop_div_kernel(float* __restrict__ acc, const float* __restrict__ cnt) {
  const int i = blockIdx.x, d = threadIdx.x;
  acc[(size_t)i * 128 + d] /= fmaxf(cnt[i], 1.f);
}

// ---------------- GAT attention ----------------
__global__ void gat_logits_kernel(const float* __restrict__ xl, const float* __restrict__ xr,
                                  const float* __restrict__ ep,
                                  const int* __restrict__ src, const int* __restrict__ dst,
                                  const float* __restrict__ att,
                                  float* __restrict__ logit, unsigned* __restrict__ mEnc, long tot)
{
  long gid = blockIdx.x * (long)blockDim.x + threadIdx.x;
  if (gid >= tot) return;
  const int h = (int)(gid & 3); const long ed = gid >> 2;
  const int s  = (ed < EEDGE) ? src[ed] : (int)(ed - EEDGE);
  const int dd = (ed < EEDGE) ? dst[ed] : (int)(ed - EEDGE);
  const float* a  = att + h * 32;
  const float* pl = xl + (size_t)s  * 128 + h * 32;
  const float* pr = xr + (size_t)dd * 128 + h * 32;
  const float* pe = ep + (size_t)ed * 128 + h * 32;
  float acc = 0.f;
  for (int c = 0; c < 32; ++c) {
    float v = pl[c] + pr[c] + pe[c];
    v = (v > 0.f) ? v : 0.2f * v;          // leaky_relu(0.2)
    acc += a[c] * v;
  }
  logit[(size_t)ed * 4 + h] = acc;
  atomicMaxF(&mEnc[(size_t)dd * 4 + h], acc);
}

__global__ void gat_exp_kernel(float* __restrict__ logit, const unsigned* __restrict__ mEnc,
                               float* __restrict__ z, const int* __restrict__ dst, long tot)
{
  long gid = blockIdx.x * (long)blockDim.x + threadIdx.x;
  if (gid >= tot) return;
  const int h = (int)(gid & 3); const long ed = gid >> 2;
  const int dd = (ed < EEDGE) ? dst[ed] : (int)(ed - EEDGE);
  const float ex = expf(logit[(size_t)ed * 4 + h] - decMaxF(mEnc[(size_t)dd * 4 + h]));
  logit[(size_t)ed * 4 + h] = ex;
  atomicAdd(&z[(size_t)dd * 4 + h], ex);
}

__global__ void gat_agg_kernel(const float* __restrict__ logit, const float* __restrict__ z,
                               const float* __restrict__ xl,
                               const int* __restrict__ src, const int* __restrict__ dst,
                               float* __restrict__ out)
{
  const int ed = blockIdx.x, d = threadIdx.x;
  const int s  = (ed < EEDGE) ? src[ed] : ed - EEDGE;
  const int dd = (ed < EEDGE) ? dst[ed] : ed - EEDGE;
  const int h  = d >> 5;
  const float alpha = logit[(size_t)ed * 4 + h] / z[(size_t)dd * 4 + h];
  atomicAdd(&out[(size_t)dd * 128 + d], alpha * xl[(size_t)s * 128 + d]);
}

// ---------------- attention (decoder) ----------------
__global__ void attn_scores_kernel(const float* __restrict__ Q, int ldq,
                                   const float* __restrict__ Kc, int ldk,
                                   float* __restrict__ P, int nq, int nk)
{
  __shared__ float sm[512];
  const int bid = blockIdx.x;
  const int qi = bid % nq;
  const int h  = (bid / nq) % HHEAD;
  const int b  = bid / (nq * HHEAD);
  const int ki = threadIdx.x;
  const float* qp = Q  + (size_t)(b * nq + qi) * ldq + h * 32;
  const float* kp = Kc + (size_t)(b * nk + ki) * ldk + h * 32;
  float s = 0.f;
  for (int d = 0; d < 32; ++d) s += qp[d] * kp[d];
  s *= 0.17677669529663687f;                          // 1/sqrt(32)
  sm[ki] = s; __syncthreads();
  for (int o = nk >> 1; o > 0; o >>= 1) { if (ki < o) sm[ki] = fmaxf(sm[ki], sm[ki + o]); __syncthreads(); }
  const float mx = sm[0]; __syncthreads();
  const float e = expf(s - mx);
  sm[ki] = e; __syncthreads();
  for (int o = nk >> 1; o > 0; o >>= 1) { if (ki < o) sm[ki] += sm[ki + o]; __syncthreads(); }
  P[(size_t)bid * nk + ki] = e / sm[0];
}

__global__ void attn_av_kernel(const float* __restrict__ P, const float* __restrict__ V, int ldv,
                               float* __restrict__ O, int nq, int nk)
{
  const int bid = blockIdx.x;
  const int qi = bid % nq, b = bid / nq;
  const int t = threadIdx.x, h = t >> 5;
  const float* pp = P + (size_t)((b * HHEAD + h) * nq + qi) * nk;
  const float* vp = V + (size_t)(b * nk) * ldv + t;
  float acc = 0.f;
  for (int k = 0; k < nk; ++k) acc += pp[k] * vp[(size_t)k * ldv];
  O[(size_t)(b * nq + qi) * 128 + t] = acc;
}

// ---------------- pooling / heads ----------------
__global__ void pool_kernel(const float* __restrict__ h, float* __restrict__ pooled) {
  const int b = blockIdx.x, d = threadIdx.x;
  const float* p = h + (size_t)b * NPGc * 128 + d;
  float s = 0.f, mx = -3.4e38f;
  for (int i = 0; i < NPGc; ++i) { float v = p[(size_t)i * 128]; s += v; mx = fmaxf(mx, v); }
  pooled[b * 256 + d]       = s * (1.f / NPGc);
  pooled[b * 256 + 128 + d] = mx;
}

__global__ void build_sf_kernel(const float* __restrict__ t, const float* __restrict__ gc,
                                float* __restrict__ sf) {
  const int r = blockIdx.x, c = threadIdx.x;   // 256 threads
  const int b = r / KQ;
  sf[(size_t)r * 256 + c] = (c < 128) ? t[(size_t)r * 128 + c] : gc[b * 128 + (c - 128)];
}

__global__ void head_hp_kernel(const float* __restrict__ X, const float* __restrict__ w,
                               const float* __restrict__ b,
                               float* __restrict__ prob, float* __restrict__ logits, int n) {
  const int r = blockIdx.x * blockDim.x + threadIdx.x;
  if (r >= n) return;
  float acc = b[0];
  const float* x = X + (size_t)r * 128;
  for (int k = 0; k < 128; ++k) acc += x[k] * w[k];
  logits[r] = acc;
  prob[r]   = 1.f / (1.f + expf(-acc));
}

__global__ void head_sp_kernel(const float* __restrict__ X, int Kd, const float* __restrict__ w,
                               const float* __restrict__ b, float* __restrict__ out,
                               float scale, float addc, int n) {
  const int r = blockIdx.x * blockDim.x + threadIdx.x;
  if (r >= n) return;
  float acc = b[0];
  const float* x = X + (size_t)r * Kd;
  for (int k = 0; k < Kd; ++k) acc += x[k] * w[k];
  out[r] = softplusf(acc) * scale + addc;
}

// ---------------------------------------------------------------------------
// host side
// ---------------------------------------------------------------------------
static inline unsigned gb(long n, int b) { return (unsigned)((n + b - 1) / b); }

static void gemm(hipStream_t s, int act,
                 const float* A, int lda, const float* W, int ldw, const float* bias,
                 float* C, int ldc, int M, int N, int K, _Float16* wsw)
{
  swizzle_w_kernel<<<gb((long)K * N, 256), 256, 0, s>>>(W, ldw, N, K, wsw);
  long waves = (long)(M / 16) * (N / 64);
  unsigned grid = gb(waves * 32, 256);
  if (act == 1)      wmma_gemm_fast<1><<<grid, 256, 0, s>>>(A, lda, wsw, bias, C, ldc, M, N, K);
  else if (act == 2) wmma_gemm_fast<2><<<grid, 256, 0, s>>>(A, lda, wsw, bias, C, ldc, M, N, K);
  else               wmma_gemm_fast<0><<<grid, 256, 0, s>>>(A, lda, wsw, bias, C, ldc, M, N, K);
}

extern "C" void kernel_launch(void* const* d_in, const int* in_sizes, int n_in,
                              void* d_out, int out_size, void* d_ws, size_t ws_size,
                              hipStream_t stream) {
  (void)in_sizes; (void)n_in; (void)out_size; (void)ws_size;
  const float* const* F = (const float* const*)d_in;

  // --- input decode: top-level insertion order; params tree-flattened with
  // --- JAX sorted-key dict order ('W' < lowercase).
  const float* x_in      = F[0];
  const float* edge_attr = F[1];
  struct DecP { const float *ca_in_b,*ca_in_w,*ca_out_b,*ca_out_w,*ff_W1,*ff_W2,*ff_b1,*ff_b2,
                            *ln1_b,*ln1_g,*ln2_b,*ln2_g,*ln3_b,*ln3_g,*sa_in_b,*sa_in_w,*sa_out_b,*sa_out_w; };
  struct GatP { const float *We,*Wl,*Wr,*att,*bias,*bl,*br,*ln_b,*ln_g; };
  int p = 2;
  DecP dec[2];
  for (int l = 0; l < 2; ++l) {
    dec[l].ca_in_b = F[p++]; dec[l].ca_in_w = F[p++]; dec[l].ca_out_b = F[p++]; dec[l].ca_out_w = F[p++];
    dec[l].ff_W1 = F[p++]; dec[l].ff_W2 = F[p++]; dec[l].ff_b1 = F[p++]; dec[l].ff_b2 = F[p++];
    dec[l].ln1_b = F[p++]; dec[l].ln1_g = F[p++]; dec[l].ln2_b = F[p++]; dec[l].ln2_g = F[p++];
    dec[l].ln3_b = F[p++]; dec[l].ln3_g = F[p++];
    dec[l].sa_in_b = F[p++]; dec[l].sa_in_w = F[p++]; dec[l].sa_out_b = F[p++]; dec[l].sa_out_w = F[p++];
  }
  const float* edge_W1 = F[p++]; const float* edge_W2 = F[p++];
  const float* edge_b1 = F[p++]; const float* edge_b2 = F[p++];
  const float* edge_beta = F[p++]; const float* edge_g = F[p++];
  GatP gat[4];
  for (int l = 0; l < 4; ++l) {
    gat[l].We = F[p++]; gat[l].Wl = F[p++]; gat[l].Wr = F[p++]; gat[l].att = F[p++];
    gat[l].bias = F[p++]; gat[l].bl = F[p++]; gat[l].br = F[p++];
    gat[l].ln_b = F[p++]; gat[l].ln_g = F[p++];
  }
  const float* gc_W = F[p++]; const float* gc_b = F[p++];
  const float* gc_beta = F[p++]; const float* gc_g = F[p++];
  const float* ha_W1 = F[p++]; const float* ha_W2 = F[p++]; const float* ha_b1 = F[p++]; const float* ha_b2 = F[p++];
  const float* hc_W1 = F[p++]; const float* hc_W2 = F[p++]; const float* hc_b1 = F[p++]; const float* hc_b2 = F[p++];
  const float* hf_W1 = F[p++]; const float* hf_W2 = F[p++]; const float* hf_b1 = F[p++]; const float* hf_b2 = F[p++];
  const float* hp_W = F[p++]; const float* hp_b = F[p++];
  const float* node_W = F[p++]; const float* node_b = F[p++];
  const float* node_beta = F[p++]; const float* node_g = F[p++];
  const float* query = F[p++];
  const float* sr_W1 = F[p++]; const float* sr_W2 = F[p++]; const float* sr_b1 = F[p++]; const float* sr_b2 = F[p++];
  const int* eidx = (const int*)d_in[p++];
  p += 2;                                  // batch, num_graphs (layout is implicit)
  const int* e_src = eidx;
  const int* e_dst = eidx + EEDGE;

  // --- workspace carve-up (floats) ---
  float* ws = (float*)d_ws;
  size_t off = 0;
  auto alloc = [&](size_t n) { float* q = ws + off; off += n; return q; };
  float* h     = alloc((size_t)NNODE * DDIM);
  float* ef    = alloc((size_t)ETOT * DDIM);
  float* ep    = alloc((size_t)ETOT * DDIM);
  float* xl    = alloc((size_t)NNODE * DDIM);
  float* xr    = alloc((size_t)NNODE * DDIM);
  float* gout  = alloc((size_t)NNODE * DDIM);
  float* tmpN  = alloc((size_t)NNODE * DDIM);
  float* elog  = alloc((size_t)ETOT * HHEAD);
  unsigned* mE = (unsigned*)alloc((size_t)NNODE * HHEAD);
  float* zbuf  = alloc((size_t)NNODE * HHEAD);
  float* cnt   = alloc(NNODE);
  float* tq    = alloc((size_t)BG * KQ * DDIM);
  float* qkv   = alloc((size_t)BG * KQ * 3 * DDIM);
  float* probs = alloc((size_t)BG * HHEAD * KQ * NPGc);
  float* aout  = alloc((size_t)BG * KQ * DDIM);
  float* tmpD  = alloc((size_t)BG * KQ * DDIM);
  float* qh    = alloc((size_t)BG * KQ * DDIM);
  float* ff1   = alloc((size_t)BG * KQ * 4 * DDIM);
  float* pooled= alloc((size_t)BG * 2 * DDIM);
  float* gc    = alloc((size_t)BG * DDIM);
  float* gch   = alloc((size_t)BG * 64);
  float* sf    = alloc((size_t)BG * KQ * 2 * DDIM);
  float* sfa   = alloc((size_t)BG * KQ * DDIM);
  float* sf2   = alloc((size_t)BG * KQ * DDIM);
  float* xpad  = alloc((size_t)NNODE * 32);     // x zero-padded K 5->32
  float* epad  = alloc((size_t)EEDGE * 32);     // edge_attr zero-padded K 4->32
  float* wpadN = alloc((size_t)32 * DDIM);      // node_W padded
  float* wpadE = alloc((size_t)32 * DDIM);      // edge_W1 padded
  _Float16* wsw = (_Float16*)alloc((size_t)32768); // 64K halves: swizzled weight staging

  const int nq = BG * KQ;                  // 2048 decoder rows
  float* out_prob   = (float*)d_out;
  float* out_logits = out_prob + nq;
  float* out_freq   = out_logits + nq;
  float* out_amp    = out_freq + nq;
  float* out_count  = out_amp + nq;

  // ================= node / edge embeddings =================
  pad_a_kernel<<<gb((long)NNODE * 32, 256), 256, 0, stream>>>(x_in, 5, xpad, NNODE);
  pad_w_kernel<<<gb(32 * DDIM, 256), 256, 0, stream>>>(node_W, 5, DDIM, wpadN);
  gemm(stream, 1, xpad, 32, wpadN, DDIM, node_b, tmpN, DDIM, NNODE, DDIM, 32, wsw);     // silu
  ln_kernel<<<NNODE, 128, 0, stream>>>(tmpN, nullptr, nullptr, node_g, node_beta, h);

  pad_a_kernel<<<gb((long)EEDGE * 32, 256), 256, 0, stream>>>(edge_attr, 4, epad, EEDGE);
  pad_w_kernel<<<gb(32 * DDIM, 256), 256, 0, stream>>>(edge_W1, 4, DDIM, wpadE);
  gemm(stream, 1, epad, 32, wpadE, DDIM, edge_b1, ep, DDIM, EEDGE, DDIM, 32, wsw);      // silu -> ep (temp)
  gemm(stream, 0, ep, DDIM, edge_W2, DDIM, edge_b2, ef, DDIM, EEDGE, DDIM, DDIM, wsw);  // -> ef[0:E)
  ln_kernel<<<EEDGE, 128, 0, stream>>>(ef, nullptr, nullptr, edge_g, edge_beta, ef);    // in-place

  // self-loop edge features (mean of incoming) -> ef[E:E+N)
  fill_u32_kernel<<<gb((long)NNODE * DDIM, 256), 256, 0, stream>>>((unsigned*)(ef + (size_t)EEDGE * DDIM), 0u, (long)NNODE * DDIM);
  fill_u32_kernel<<<gb(NNODE, 256), 256, 0, stream>>>((unsigned*)cnt, 0u, NNODE);
  eloop_acc_kernel<<<EEDGE, 128, 0, stream>>>(ef, e_dst, ef + (size_t)EEDGE * DDIM, cnt);
  eloop_div_kernel<<<NNODE, 128, 0, stream>>>(ef + (size_t)EEDGE * DDIM, cnt);

  // ================= 4 GATv2 layers =================
  for (int l = 0; l < 4; ++l) {
    const GatP& g = gat[l];
    gemm(stream, 0, h, DDIM, g.Wl, DDIM, g.bl, xl, DDIM, NNODE, DDIM, DDIM, wsw);
    gemm(stream, 0, h, DDIM, g.Wr, DDIM, g.br, xr, DDIM, NNODE, DDIM, DDIM, wsw);
    gemm(stream, 0, ef, DDIM, g.We, DDIM, nullptr, ep, DDIM, ETOT, DDIM, DDIM, wsw);

    fill_u32_kernel<<<gb((long)NNODE * HHEAD, 256), 256, 0, stream>>>(mE, 0x007FFFFFu, (long)NNODE * HHEAD); // enc(-inf)
    gat_logits_kernel<<<gb((long)ETOT * HHEAD, 256), 256, 0, stream>>>(xl, xr, ep, e_src, e_dst, g.att, elog, mE, (long)ETOT * HHEAD);
    fill_u32_kernel<<<gb((long)NNODE * HHEAD, 256), 256, 0, stream>>>((unsigned*)zbuf, 0u, (long)NNODE * HHEAD);
    gat_exp_kernel<<<gb((long)ETOT * HHEAD, 256), 256, 0, stream>>>(elog, mE, zbuf, e_dst, (long)ETOT * HHEAD);
    fill_u32_kernel<<<gb((long)NNODE * DDIM, 256), 256, 0, stream>>>((unsigned*)gout, 0u, (long)NNODE * DDIM);
    gat_agg_kernel<<<ETOT, 128, 0, stream>>>(elog, zbuf, xl, e_src, e_dst, gout);

    ln_kernel<<<NNODE, 128, 0, stream>>>(gout, nullptr, g.bias, g.ln_g, g.ln_b, tmpN);
    gelu_add_kernel<<<gb((long)NNODE * DDIM, 256), 256, 0, stream>>>(h, tmpN, (long)NNODE * DDIM);
  }

  // ================= transformer decoder (mem = h, dense [B,512,128]) =================
  bcast_query_kernel<<<gb((long)nq * DDIM, 256), 256, 0, stream>>>(query, tq, (long)nq * DDIM);
  for (int l = 0; l < 2; ++l) {
    const DecP& d = dec[l];
    // self attention
    gemm(stream, 0, tq, DDIM, d.sa_in_w, 3 * DDIM, d.sa_in_b, qkv, 3 * DDIM, nq, 3 * DDIM, DDIM, wsw);
    attn_scores_kernel<<<BG * HHEAD * KQ, KQ, 0, stream>>>(qkv, 3 * DDIM, qkv + DDIM, 3 * DDIM, probs, KQ, KQ);
    attn_av_kernel<<<nq, 128, 0, stream>>>(probs, qkv + 2 * DDIM, 3 * DDIM, aout, KQ, KQ);
    gemm(stream, 0, aout, DDIM, d.sa_out_w, DDIM, d.sa_out_b, tmpD, DDIM, nq, DDIM, DDIM, wsw);
    ln_kernel<<<nq, 128, 0, stream>>>(tmpD, tq, nullptr, d.ln1_g, d.ln1_b, tq);
    // cross attention (k/v from node memory)
    gemm(stream, 0, tq, DDIM, d.ca_in_w,            3 * DDIM, d.ca_in_b,            qh, DDIM, nq,    DDIM, DDIM, wsw);
    gemm(stream, 0, h,  DDIM, d.ca_in_w + DDIM,     3 * DDIM, d.ca_in_b + DDIM,     xl, DDIM, NNODE, DDIM, DDIM, wsw);
    gemm(stream, 0, h,  DDIM, d.ca_in_w + 2 * DDIM, 3 * DDIM, d.ca_in_b + 2 * DDIM, xr, DDIM, NNODE, DDIM, DDIM, wsw);
    attn_scores_kernel<<<BG * HHEAD * KQ, NPGc, 0, stream>>>(qh, DDIM, xl, DDIM, probs, KQ, NPGc);
    attn_av_kernel<<<nq, 128, 0, stream>>>(probs, xr, DDIM, aout, KQ, NPGc);
    gemm(stream, 0, aout, DDIM, d.ca_out_w, DDIM, d.ca_out_b, tmpD, DDIM, nq, DDIM, DDIM, wsw);
    ln_kernel<<<nq, 128, 0, stream>>>(tmpD, tq, nullptr, d.ln2_g, d.ln2_b, tq);
    // feed-forward
    gemm(stream, 2, tq, DDIM, d.ff_W1, 4 * DDIM, d.ff_b1, ff1, 4 * DDIM, nq, 4 * DDIM, DDIM, wsw); // gelu
    gemm(stream, 0, ff1, 4 * DDIM, d.ff_W2, DDIM, d.ff_b2, tmpD, DDIM, nq, DDIM, 4 * DDIM, wsw);
    ln_kernel<<<nq, 128, 0, stream>>>(tmpD, tq, nullptr, d.ln3_g, d.ln3_b, tq);
  }

  // ================= pooling + global context =================
  pool_kernel<<<BG, 128, 0, stream>>>(h, pooled);
  gemm(stream, 1, pooled, 2 * DDIM, gc_W, DDIM, gc_b, tmpD, DDIM, BG, DDIM, 2 * DDIM, wsw);     // silu
  ln_kernel<<<BG, 128, 0, stream>>>(tmpD, nullptr, nullptr, gc_g, gc_beta, gc);

  // ================= spectral refine + heads =================
  build_sf_kernel<<<nq, 256, 0, stream>>>(tq, gc, sf);
  gemm(stream, 2, sf, 2 * DDIM, sr_W1, DDIM, sr_b1, sfa, DDIM, nq, DDIM, 2 * DDIM, wsw);        // gelu
  gemm(stream, 0, sfa, DDIM, sr_W2, DDIM, sr_b2, sf2, DDIM, nq, DDIM, DDIM, wsw);

  head_hp_kernel<<<gb(nq, 256), 256, 0, stream>>>(sf2, hp_W, hp_b, out_prob, out_logits, nq);

  gemm(stream, 1, sf2, DDIM, hf_W1, DDIM, hf_b1, sfa, DDIM, nq, DDIM, DDIM, wsw);               // silu
  head_sp_kernel<<<gb(nq, 256), 256, 0, stream>>>(sfa, 128, hf_W2, hf_b2, out_freq, 1.f, 1e-5f, nq);

  gemm(stream, 1, sf2, DDIM, ha_W1, DDIM, ha_b1, sfa, DDIM, nq, DDIM, DDIM, wsw);               // silu
  head_sp_kernel<<<gb(nq, 256), 256, 0, stream>>>(sfa, 128, ha_W2, ha_b2, out_amp, 1e-3f, 0.f, nq);

  gemm(stream, 1, gc, DDIM, hc_W1, 64, hc_b1, gch, 64, BG, 64, DDIM, wsw);                      // silu
  head_sp_kernel<<<1, 256, 0, stream>>>(gch, 64, hc_W2, hc_b2, out_count, 1.f, 0.f, BG);
}
